// RelPartialLearnableMultiHeadAttn_32452772888670
// MI455X (gfx1250) — compile-verified, matrix operand order checked by reading, not run on hardware
//
#include <hip/hip_runtime.h>
#include <hip/hip_bf16.h>

#define N_HEADS 16
#define D_HEADD 64
#define D_MODELD 1024
#define QLEND 1024
#define BSZD 4
#define CONVK 7
#define SCALEF 0.125f

typedef __bf16 bf16_t;
typedef __attribute__((ext_vector_type(16))) bf16_t v16bf;
typedef __attribute__((ext_vector_type(8)))  bf16_t v8bf;
typedef __attribute__((ext_vector_type(8)))  float  v8f;

static __device__ inline v8f wmma_bf16(v16bf a, v16bf b, v8f c) {
  // D = A(16x32) * B(32x16) + C, f32 accumulate
  return __builtin_amdgcn_wmma_f32_16x16x32_bf16(false, a, false, b, (short)0, c, false, false);
}

// A/B fragment from bf16 row-major [rows, ldk]: lane -> row = base_row + (lane&15),
// element e -> k = k0 + e + (e>=8 ? 8 : 0) + (lane>=16 ? 8 : 0)  (per CDNA5 16-bit layout)
static __device__ inline v16bf load_frag_bf16(const bf16_t* __restrict__ base,
                                              int ldk, int row, int k0, int lane) {
  const bf16_t* p = base + (size_t)row * ldk + k0 + ((lane & 16) ? 8 : 0);
  v8bf lo = *(const v8bf*)(p);
  v8bf hi = *(const v8bf*)(p + 16);
  v16bf f;
#pragma unroll
  for (int e = 0; e < 8; ++e) { f[e] = lo[e]; f[e + 8] = hi[e]; }
  return f;
}

// ---------------- f32 -> bf16 convert ----------------
__global__ void cvt_bf16_kernel(const float* __restrict__ src, bf16_t* __restrict__ dst, int n) {
  int i = blockIdx.x * blockDim.x + threadIdx.x;
  if (i < n) dst[i] = (bf16_t)src[i];
}

// ---------------- generic NT GEMM: C[M,N] = A[M,K] * B[N,K]^T ----------------
// block = 128 threads (4 waves), block tile 64(M) x 64(N); wave w -> rows [m0+16w, +16)
__global__ __launch_bounds__(128) void gemm_nt_bf16(const bf16_t* __restrict__ A,
                                                    const bf16_t* __restrict__ B,
                                                    float* __restrict__ C,
                                                    int M, int N, int K) {
  const int lane = threadIdx.x & 31;
  const int wave = threadIdx.x >> 5;
  const int n0 = blockIdx.x * 64;
  const int m0 = blockIdx.y * 64 + wave * 16;
  const int arow = m0 + (lane & 15);
  v8f acc0 = {}, acc1 = {}, acc2 = {}, acc3 = {};
  for (int k0 = 0; k0 < K; k0 += 32) {
    v16bf a = load_frag_bf16(A, K, arow, k0, lane);
    v16bf b0 = load_frag_bf16(B, K, n0 + 0  + (lane & 15), k0, lane);
    v16bf b1 = load_frag_bf16(B, K, n0 + 16 + (lane & 15), k0, lane);
    v16bf b2 = load_frag_bf16(B, K, n0 + 32 + (lane & 15), k0, lane);
    v16bf b3 = load_frag_bf16(B, K, n0 + 48 + (lane & 15), k0, lane);
    acc0 = wmma_bf16(a, b0, acc0);
    acc1 = wmma_bf16(a, b1, acc1);
    acc2 = wmma_bf16(a, b2, acc2);
    acc3 = wmma_bf16(a, b3, acc3);
  }
  const int rbase = m0 + ((lane & 16) ? 8 : 0);
  const int cbase = n0 + (lane & 15);
#pragma unroll
  for (int r = 0; r < 8; ++r) {
    size_t ro = (size_t)(rbase + r) * N;
    C[ro + cbase +  0] = acc0[r];
    C[ro + cbase + 16] = acc1[r];
    C[ro + cbase + 32] = acc2[r];
    C[ro + cbase + 48] = acc3[r];
  }
}

// ---------------- conv1d over sequence, per (head,batch) channel block ----------------
// heads: [L, BSZ, 3*1024]; slice sel picks q/k/v. out: [L, BSZ, 1024] (n*64+dout)
__global__ __launch_bounds__(64) void conv1d_kernel(const float* __restrict__ heads,
                                                    const float* __restrict__ wgt,
                                                    const float* __restrict__ bias,
                                                    float* __restrict__ out, int sel) {
  const int dout = threadIdx.x;        // 0..63
  const int L    = blockIdx.x;         // 0..QLEN-1
  const int n    = blockIdx.y / BSZD;
  const int b    = blockIdx.y % BSZD;
  float acc = bias[dout];
  const float* wbase = wgt + dout * D_HEADD * CONVK;
#pragma unroll
  for (int t = 0; t < CONVK; ++t) {
    int Ls = L + t - CONVK / 2;
    if (Ls < 0 || Ls >= QLEND) continue;
    const float* x = heads + ((size_t)Ls * BSZD + b) * (3 * D_MODELD) + sel * D_MODELD + n * D_HEADD;
#pragma unroll
    for (int din = 0; din < D_HEADD; ++din)
      acc += x[din] * wbase[din * CONVK + t];
  }
  out[((size_t)L * BSZD + b) * D_MODELD + n * D_HEADD + dout] = acc;
}

// ---------------- attention: one block per (b, n, 16-row q strip) ----------------
// S strip 16 x 1024 f32 in LDS (64KB). AC + shifted BD + softmax + P*V, all WMMA.
__global__ __launch_bounds__(128) void attn_kernel(const float* __restrict__ q_h,
                                                   const float* __restrict__ k_h,
                                                   const float* __restrict__ v_h,
                                                   const float* __restrict__ rk,
                                                   const float* __restrict__ rwb,
                                                   const float* __restrict__ rrb,
                                                   float* __restrict__ vec) {
  __shared__ float S[16 * 1024];
  const int lane = threadIdx.x & 31;
  const int wave = threadIdx.x >> 5;
  const int i0 = blockIdx.x * 16;
  const int b  = blockIdx.y;
  const int n  = blockIdx.z;
  const int hi8  = (lane & 16) ? 8 : 0;
  const int lo16 = lane & 15;
  const int mrow = i0 + lo16;

  // Q fragments (K-dim = 64 -> 2 steps) with both relative biases, f32 -> bf16
  v16bf qw[2], qr[2];
#pragma unroll
  for (int ks = 0; ks < 2; ++ks) {
#pragma unroll
    for (int e = 0; e < 16; ++e) {
      int d = ks * 32 + e + ((e >= 8) ? 8 : 0) + hi8;
      float qv = q_h[((size_t)mrow * BSZD + b) * D_MODELD + n * D_HEADD + d];
      qw[ks][e] = (bf16_t)(qv + rwb[n * D_HEADD + d]);
      qr[ks][e] = (bf16_t)(qv + rrb[n * D_HEADD + d]);
    }
  }

  const int ntiles = blockIdx.x + 1;      // 16-wide column tiles: j in [0, 16*ntiles)
  const int ncols  = 16 * ntiles;

  // ---- AC = (q + r_w_bias) . k^T ----
  for (int t = wave; t < ntiles; t += 4) {
    const int j0 = t * 16;
    v8f acc = {};
#pragma unroll
    for (int ks = 0; ks < 2; ++ks) {
      v16bf bf;
      const int jr = j0 + lo16;
#pragma unroll
      for (int e = 0; e < 16; ++e) {
        int d = ks * 32 + e + ((e >= 8) ? 8 : 0) + hi8;
        bf[e] = (bf16_t)k_h[((size_t)jr * BSZD + b) * D_MODELD + n * D_HEADD + d];
      }
      acc = wmma_bf16(qw[ks], bf, acc);
    }
#pragma unroll
    for (int r = 0; r < 8; ++r)
      S[(r + hi8) * 1024 + j0 + lo16] = acc[r];
  }
  __syncthreads();

  // ---- BD band: BDraw[i, c], c in [L-16-i0, L); rel_shift: j = c - (L-1) + i ----
  const int clo = QLEND - 16 - i0;
  for (int t = wave; t < ntiles; t += 4) {
    const int c0 = clo + t * 16;
    v8f acc = {};
#pragma unroll
    for (int ks = 0; ks < 2; ++ks) {
      v16bf bf;
      const int cr = c0 + lo16;
#pragma unroll
      for (int e = 0; e < 16; ++e) {
        int d = ks * 32 + e + ((e >= 8) ? 8 : 0) + hi8;
        bf[e] = (bf16_t)rk[((size_t)cr * N_HEADS + n) * D_HEADD + d];
      }
      acc = wmma_bf16(qr[ks], bf, acc);
    }
    const int c = c0 + lo16;
#pragma unroll
    for (int r = 0; r < 8; ++r) {
      int di = r + hi8;
      int j = c - (QLEND - 1) + (i0 + di);
      if (j >= 0) S[di * 1024 + j] += acc[r];
    }
  }
  __syncthreads();

  // ---- masked softmax over j in [0, i] (wave owns 4 rows, 8 lanes per row) ----
  {
    const int srow = wave * 4 + (lane >> 3);
    const int seg  = lane & 7;
    const int irow = i0 + srow;
    float m = -3.0e38f;
    for (int j = seg; j <= irow; j += 8)
      m = fmaxf(m, S[srow * 1024 + j] * SCALEF);
#pragma unroll
    for (int off = 1; off < 8; off <<= 1) m = fmaxf(m, __shfl_xor(m, off, 8));
    float sum = 0.f;
    for (int j = seg; j <= irow; j += 8) {
      float p = __expf(S[srow * 1024 + j] * SCALEF - m);
      S[srow * 1024 + j] = p;
      sum += p;
    }
#pragma unroll
    for (int off = 1; off < 8; off <<= 1) sum += __shfl_xor(sum, off, 8);
    const float inv = 1.0f / sum;
    for (int j = seg; j <= irow; j += 8) S[srow * 1024 + j] *= inv;
    const int ncols32 = (ncols + 31) & ~31;
    for (int j = irow + 1 + seg; j < ncols32; j += 8) S[srow * 1024 + j] = 0.f;
  }
  __syncthreads();

  // ---- O = P * V : wave w computes 16 rows x 16 cols (d = 16w + lane&15) ----
  {
    const int ncols32 = (ncols + 31) & ~31;
    const int dcol = wave * 16 + lo16;
    v8f oacc = {};
    for (int k0 = 0; k0 < ncols32; k0 += 32) {
      v16bf pa, vb;
#pragma unroll
      for (int e = 0; e < 16; ++e) {
        int j = k0 + e + ((e >= 8) ? 8 : 0) + hi8;
        pa[e] = (bf16_t)S[lo16 * 1024 + j];
        vb[e] = (bf16_t)v_h[((size_t)j * BSZD + b) * D_MODELD + n * D_HEADD + dcol];
      }
      oacc = wmma_bf16(pa, vb, oacc);
    }
#pragma unroll
    for (int r = 0; r < 8; ++r) {
      int i = i0 + r + hi8;
      vec[((size_t)i * BSZD + b) * D_MODELD + n * D_HEADD + dcol] = oacc[r];
    }
  }
}

// ---------------- residual + LayerNorm ----------------
__global__ __launch_bounds__(256) void ln_kernel(const float* __restrict__ w,
                                                 const float* __restrict__ attn,
                                                 const float* __restrict__ gamma,
                                                 const float* __restrict__ beta,
                                                 float* __restrict__ out) {
  __shared__ float red[256];
  const int row = blockIdx.x;
  const int tid = threadIdx.x;
  const float* wr = w + (size_t)row * D_MODELD;
  const float* ar = attn + (size_t)row * D_MODELD;
  float x[4];
  float lsum = 0.f;
#pragma unroll
  for (int k = 0; k < 4; ++k) {
    x[k] = wr[tid + 256 * k] + ar[tid + 256 * k];
    lsum += x[k];
  }
  red[tid] = lsum; __syncthreads();
  for (int s = 128; s > 0; s >>= 1) { if (tid < s) red[tid] += red[tid + s]; __syncthreads(); }
  const float mu = red[0] * (1.0f / D_MODELD);
  __syncthreads();
  float lvar = 0.f;
#pragma unroll
  for (int k = 0; k < 4; ++k) { float dx = x[k] - mu; lvar += dx * dx; }
  red[tid] = lvar; __syncthreads();
  for (int s = 128; s > 0; s >>= 1) { if (tid < s) red[tid] += red[tid + s]; __syncthreads(); }
  const float rstd = rsqrtf(red[0] * (1.0f / D_MODELD) + 1e-5f);
#pragma unroll
  for (int k = 0; k < 4; ++k) {
    int c = tid + 256 * k;
    out[(size_t)row * D_MODELD + c] = (x[k] - mu) * rstd * gamma[c] + beta[c];
  }
}

extern "C" void kernel_launch(void* const* d_in, const int* in_sizes, int n_in,
                              void* d_out, int out_size, void* d_ws, size_t ws_size,
                              hipStream_t stream) {
  (void)in_sizes; (void)n_in; (void)out_size; (void)ws_size;
  const float* w    = (const float*)d_in[0];
  const float* r    = (const float*)d_in[1];
  const float* rwb  = (const float*)d_in[2];
  const float* rrb  = (const float*)d_in[3];
  const float* Wqkv = (const float*)d_in[4];
  const float* Wr   = (const float*)d_in[5];
  const float* Wo   = (const float*)d_in[6];
  const float* cwq  = (const float*)d_in[7];
  const float* cbq  = (const float*)d_in[8];
  const float* cwk  = (const float*)d_in[9];
  const float* cbk  = (const float*)d_in[10];
  const float* cwv  = (const float*)d_in[11];
  const float* cbv  = (const float*)d_in[12];
  const float* gam  = (const float*)d_in[13];
  const float* bet  = (const float*)d_in[14];
  float* out = (float*)d_out;

  const int MQ = QLEND * BSZD;            // 4096
  char* ws = (char*)d_ws;
  auto take = [&](size_t bytes) { char* p = ws; ws += (bytes + 255) & ~(size_t)255; return p; };
  float*  heads   = (float*)take((size_t)MQ * 3 * D_MODELD * 4);
  float*  q_h     = (float*)take((size_t)MQ * D_MODELD * 4);
  float*  k_h     = (float*)take((size_t)MQ * D_MODELD * 4);
  float*  v_h     = (float*)take((size_t)MQ * D_MODELD * 4);
  float*  rkf     = (float*)take((size_t)QLEND * D_MODELD * 4);
  float*  vec     = (float*)take((size_t)MQ * D_MODELD * 4);
  float*  attn    = (float*)take((size_t)MQ * D_MODELD * 4);
  bf16_t* w_bf    = (bf16_t*)take((size_t)MQ * D_MODELD * 2);
  bf16_t* Wqkv_bf = (bf16_t*)take((size_t)3 * D_MODELD * D_MODELD * 2);
  bf16_t* r_bf    = (bf16_t*)take((size_t)QLEND * D_MODELD * 2);
  bf16_t* Wr_bf   = (bf16_t*)take((size_t)D_MODELD * D_MODELD * 2);
  bf16_t* Wo_bf   = (bf16_t*)take((size_t)D_MODELD * D_MODELD * 2);
  bf16_t* vec_bf  = (bf16_t*)take((size_t)MQ * D_MODELD * 2);

  auto cvt = [&](const float* s, bf16_t* d, int n) {
    cvt_bf16_kernel<<<(n + 255) / 256, 256, 0, stream>>>(s, d, n);
  };
  cvt(w, w_bf, MQ * D_MODELD);
  cvt(Wqkv, Wqkv_bf, 3 * D_MODELD * D_MODELD);
  cvt(r, r_bf, QLEND * D_MODELD);
  cvt(Wr, Wr_bf, D_MODELD * D_MODELD);
  cvt(Wo, Wo_bf, D_MODELD * D_MODELD);

  // heads = w @ W_qkv^T  : [4096, 3072]
  gemm_nt_bf16<<<dim3(3 * D_MODELD / 64, MQ / 64), 128, 0, stream>>>(
      w_bf, Wqkv_bf, heads, MQ, 3 * D_MODELD, D_MODELD);
  // r_head_k = r @ W_r^T : [1024, 1024] = [rlen, n*d]
  gemm_nt_bf16<<<dim3(D_MODELD / 64, QLEND / 64), 128, 0, stream>>>(
      r_bf, Wr_bf, rkf, QLEND, D_MODELD, D_MODELD);

  // conv1d on q/k/v head slices
  dim3 cgrid(QLEND, N_HEADS * BSZD);
  conv1d_kernel<<<cgrid, 64, 0, stream>>>(heads, cwq, cbq, q_h, 0);
  conv1d_kernel<<<cgrid, 64, 0, stream>>>(heads, cwk, cbk, k_h, 1);
  conv1d_kernel<<<cgrid, 64, 0, stream>>>(heads, cwv, cbv, v_h, 2);

  // attention (AC + rel-shifted BD + softmax + P*V)
  attn_kernel<<<dim3(QLEND / 16, BSZD, N_HEADS), 128, 0, stream>>>(
      q_h, k_h, v_h, rkf, rwb, rrb, vec);

  // attn_out = vec @ W_o^T
  cvt(vec, vec_bf, MQ * D_MODELD);
  gemm_nt_bf16<<<dim3(D_MODELD / 64, MQ / 64), 128, 0, stream>>>(
      vec_bf, Wo_bf, attn, MQ, D_MODELD, D_MODELD);

  // residual + LayerNorm -> d_out
  ln_kernel<<<MQ, 256, 0, stream>>>(w, attn, gam, bet, out);
}
